// FlowEvolve_59777354825821
// MI455X (gfx1250) — compile-verified, matrix operand order checked by reading, not run on hardware
//
#include <hip/hip_runtime.h>
#include <hip/hip_bf16.h>

// ---------------- problem constants ----------------
constexpr int NF  = 20000;
constexpr int NM  = 5000;
constexpr int EFF = 100000;
constexpr int EMF = 80000;

typedef __bf16 bf16_t;
typedef bf16_t v16bf __attribute__((ext_vector_type(16)));
typedef bf16_t v8bf  __attribute__((ext_vector_type(8)));
typedef float  v8f   __attribute__((ext_vector_type(8)));
typedef unsigned int u32x4 __attribute__((ext_vector_type(4)));
typedef int i32x8 __attribute__((ext_vector_type(8)));
typedef int i32x4 __attribute__((ext_vector_type(4)));

// 16x32 bf16 fragment: lane's 16 elements are K = {base..base+7} and {base+16..base+23},
// base = (lane>=16 ? 8 : 0) (+ chunk offset). Two contiguous 8-element runs -> 2x ds_load_b128.
__device__ __forceinline__ v16bf frag_ld(const bf16_t* p) {
  v8bf lo = *(const v8bf*)p;
  v8bf hi = *(const v8bf*)(p + 16);
  return __builtin_shufflevector(lo, hi, 0, 1, 2, 3, 4, 5, 6, 7, 8, 9, 10, 11, 12, 13, 14, 15);
}

// ---------------- conversion / transpose kernels (run once per call) ----------------
__global__ void k_cvt_t512(const float* __restrict__ in, bf16_t* __restrict__ out) {
  int i = blockIdx.x * 256 + threadIdx.x;           // Wk1 (512x512) -> Wk1^T bf16
  if (i < 512 * 512) { int k = i >> 9, n = i & 511; out[(size_t)n * 512 + k] = (bf16_t)in[i]; }
}
__global__ void k_cvt_tk2(const float* __restrict__ in, bf16_t* __restrict__ out) {
  int i = blockIdx.x * 256 + threadIdx.x;           // Wk2 (512x(64x64)) -> per-k [f][d] bf16
  if (i < 512 * 4096) {
    int k = i >> 12, r = i & 4095, d = r >> 6, f = r & 63;
    out[(size_t)k * 4096 + f * 64 + d] = (bf16_t)in[i];
  }
}
__global__ void k_cvt_tb2(const float* __restrict__ bk2, bf16_t* __restrict__ out) {
  int i = blockIdx.x * 256 + threadIdx.x;           // bk2 -> row 512, [f][d]
  if (i < 4096) { int d = i >> 6, f = i & 63; out[(size_t)512 * 4096 + f * 64 + d] = (bf16_t)bk2[i]; }
}
__global__ void k_cvt_t64(const float* __restrict__ in, bf16_t* __restrict__ out) {
  int i = blockIdx.x * 256 + threadIdx.x;           // 64x64 W -> W^T bf16
  if (i < 4096) { int k = i >> 6, n = i & 63; out[n * 64 + k] = (bf16_t)in[i]; }
}
__global__ void k_zero_f32(float* __restrict__ p, int n) {
  int i = blockIdx.x * 256 + threadIdx.x;
  if (i < n) p[i] = 0.0f;
}

// tau -> temb (silu) -> ss = temb @ Wt2 + bt2, one block
__global__ void k_temb(const float* __restrict__ tau, const float* __restrict__ Wt1,
                       const float* __restrict__ bt1, const float* __restrict__ Wt2,
                       const float* __restrict__ bt2, float* __restrict__ ss) {
  __shared__ float te[64];
  int t = threadIdx.x;
  if (t < 64) {
    float x = tau[0] * Wt1[t] + bt1[t];
    te[t] = x / (1.0f + __expf(-x));
  }
  __syncthreads();
  if (t < 128) {
    float a = bt2[t];
    for (int d = 0; d < 64; ++d) a += te[d] * Wt2[d * 128 + t];
    ss[t] = a;
  }
}

// ---------------- WMMA GEMM: C[M x 64] = A[M x 64] (f32) @ B (64x64, pre-transposed bf16) ----------
__global__ __launch_bounds__(256) void k_gemm64_wmma(const float* __restrict__ Ag,
                                                     const bf16_t* __restrict__ BTg,
                                                     float* __restrict__ Cg, int M) {
  __shared__ __align__(16) bf16_t As[128 * 64];
  __shared__ __align__(16) bf16_t Bs[64 * 64];      // B^T: [n][k]
  const int tid = threadIdx.x;
  const int lane = tid & 31, wid = tid >> 5;
  const int hh = lane >> 4, r15 = lane & 15;
  const int m0 = blockIdx.x * 128;

#pragma unroll
  for (int t = 0; t < 4; ++t) {                     // stage A: 8 elems/thread/iter, vector ld/st
    int base = t * 2048 + tid * 8;
    int row = base >> 6, col = base & 63;
    int gr = m0 + row;
    v8bf pk;
    if (gr < M) {
      const float4 p0 = *(const float4*)(Ag + (size_t)gr * 64 + col);
      const float4 p1 = *(const float4*)(Ag + (size_t)gr * 64 + col + 4);
      pk[0] = (bf16_t)p0.x; pk[1] = (bf16_t)p0.y; pk[2] = (bf16_t)p0.z; pk[3] = (bf16_t)p0.w;
      pk[4] = (bf16_t)p1.x; pk[5] = (bf16_t)p1.y; pk[6] = (bf16_t)p1.z; pk[7] = (bf16_t)p1.w;
    } else {
#pragma unroll
      for (int i = 0; i < 8; ++i) pk[i] = (bf16_t)0.0f;
    }
    *(v8bf*)&As[base] = pk;
  }
#pragma unroll
  for (int t = 0; t < 2; ++t) {                     // stage B^T (uint4 copies)
    int idx = t * 256 + tid;
    ((uint4*)Bs)[idx] = ((const uint4*)BTg)[idx];
  }
  __syncthreads();

  v8f acc[4];
#pragma unroll
  for (int nt = 0; nt < 4; ++nt)
#pragma unroll
    for (int j = 0; j < 8; ++j) acc[nt][j] = 0.0f;

  const int arow = wid * 16 + r15;
#pragma unroll
  for (int kc = 0; kc < 64; kc += 32) {
    const v16bf af = frag_ld(&As[arow * 64 + kc + hh * 8]);
#pragma unroll
    for (int nt = 0; nt < 4; ++nt) {
      const v16bf bfr = frag_ld(&Bs[(nt * 16 + r15) * 64 + kc + hh * 8]);
      acc[nt] = __builtin_amdgcn_wmma_f32_16x16x32_bf16(false, af, false, bfr, (short)0,
                                                        acc[nt], false, false);
    }
  }
#pragma unroll
  for (int nt = 0; nt < 4; ++nt)
#pragma unroll
    for (int j = 0; j < 8; ++j) {
      int mrow = m0 + wid * 16 + j + hh * 8;
      if (mrow < M) Cg[(size_t)mrow * 64 + nt * 16 + r15] = acc[nt][j];
    }
}

// ---------------- fused MLP: h1 = relu(relu(ea@Wk0+bk0) @ Wk1 + bk1), bf16 out ----------------
__global__ __launch_bounds__(256) void k_h1_wmma(const float* __restrict__ ea_ff,
                                                 const float* __restrict__ Wk0,
                                                 const float* __restrict__ bk0,
                                                 const bf16_t* __restrict__ Wk1T,
                                                 const float* __restrict__ bk1,
                                                 bf16_t* __restrict__ h1) {
  __shared__ float eas[128 * 8];
  __shared__ __align__(16) bf16_t As[128 * 64];
  __shared__ __align__(16) bf16_t Bs[64 * 64];      // Wk1^T chunk: [n][k]
  const int tid = threadIdx.x;
  const int lane = tid & 31, wid = tid >> 5;
  const int hh = lane >> 4, r15 = lane & 15;
  const int m0 = blockIdx.x * 128;
  const int n0 = blockIdx.y * 64;

#pragma unroll
  for (int t = 0; t < 4; ++t) {
    int idx = t * 256 + tid;
    int row = idx >> 3, j = idx & 7;
    int gr = m0 + row;
    eas[idx] = (j < 6 && gr < EFF) ? ea_ff[(size_t)gr * 6 + j] : 0.0f;
  }
  __syncthreads();

  v8f acc[4];
#pragma unroll
  for (int nt = 0; nt < 4; ++nt)
#pragma unroll
    for (int j = 0; j < 8; ++j) acc[nt][j] = 0.0f;

  const int arow = wid * 16 + r15;
  for (int kc0 = 0; kc0 < 512; kc0 += 64) {
    __syncthreads();
#pragma unroll
    for (int t = 0; t < 4; ++t) {                   // recompute h0 chunk, 8 outputs/thread/iter
      int base = t * 2048 + tid * 8;
      int row = base >> 6, col = base & 63;
      v8bf pk;
#pragma unroll
      for (int j8 = 0; j8 < 8; ++j8) {
        int kk = kc0 + col + j8;
        float h = bk0[kk];
#pragma unroll
        for (int j = 0; j < 6; ++j) h += eas[row * 8 + j] * Wk0[j * 512 + kk];
        pk[j8] = (bf16_t)fmaxf(h, 0.0f);
      }
      *(v8bf*)&As[base] = pk;
    }
#pragma unroll
    for (int t = 0; t < 2; ++t) {                   // stage Wk1^T chunk
      int idx4 = t * 256 + tid;
      int n = idx4 >> 3, c4 = idx4 & 7;
      ((uint4*)Bs)[idx4] = *(const uint4*)(Wk1T + (size_t)(n0 + n) * 512 + kc0 + c4 * 8);
    }
    __syncthreads();
#pragma unroll
    for (int kc = 0; kc < 64; kc += 32) {
      const v16bf af = frag_ld(&As[arow * 64 + kc + hh * 8]);
#pragma unroll
      for (int nt = 0; nt < 4; ++nt) {
        const v16bf bfr = frag_ld(&Bs[(nt * 16 + r15) * 64 + kc + hh * 8]);
        acc[nt] = __builtin_amdgcn_wmma_f32_16x16x32_bf16(false, af, false, bfr, (short)0,
                                                          acc[nt], false, false);
      }
    }
  }
#pragma unroll
  for (int nt = 0; nt < 4; ++nt)
#pragma unroll
    for (int j = 0; j < 8; ++j) {
      int mrow = m0 + wid * 16 + j + hh * 8;
      if (mrow < EFF) {
        float v = acc[nt][j] + bk1[n0 + nt * 16 + r15];
        h1[(size_t)mrow * 512 + n0 + nt * 16 + r15] = (bf16_t)fmaxf(v, 0.0f);
      }
    }
}

// ---------------- big fused GEMM: msg[e,f] = sum_{k,d} h1[e,k]*x[src,d]*Wk2[k,d,f] (+ x·bk2) -------
// Output tile is (f x e): A = Wk2^T slice from LDS (2x ds_load_b128/frag), B = h⊗x built in VGPRs.
__global__ __launch_bounds__(128) void k_msg_wmma(const float* __restrict__ x_flow,
                                                  const int* __restrict__ ei_ff,
                                                  const bf16_t* __restrict__ Wk2T,
                                                  const bf16_t* __restrict__ h1,
                                                  float* __restrict__ aggr) {
  __shared__ __align__(16) bf16_t hs[64 * 64];      // 64 edges x 64-k chunk of h1
  __shared__ __align__(16) bf16_t bs[64 * 64];      // Wk2^T slice for one k: [f][d]
  const int tid = threadIdx.x;                      // 128 threads = 4 waves, 16 edges per wave
  const int lane = tid & 31, wid = tid >> 5;
  const int hh = lane >> 4, r15 = lane & 15;
  const int e0 = blockIdx.x * 64;
  const int e0w = e0 + wid * 16;

  // this lane's edge (one edge per lane column in the output)
  const int ge = e0w + r15;
  const int sidx = (ge < EFF) ? ei_ff[ge] : 0;
  const int didx = (ge < EFF) ? ei_ff[EFF + ge] : -1;

  // x values for this edge: d in groups {c*16 + hh*8 .. +8}, c = 0..3
  float xa[32];
#pragma unroll
  for (int c = 0; c < 4; ++c)
#pragma unroll
    for (int q2 = 0; q2 < 2; ++q2) {
      const float4 p = *(const float4*)(x_flow + (size_t)sidx * 64 + c * 16 + hh * 8 + q2 * 4);
      xa[c * 8 + q2 * 4 + 0] = p.x;
      xa[c * 8 + q2 * 4 + 1] = p.y;
      xa[c * 8 + q2 * 4 + 2] = p.z;
      xa[c * 8 + q2 * 4 + 3] = p.w;
    }

  v8f acc[4];                                       // acc[ft][j] = msg[f = ft*16+j+hh*8][e = r15]
#pragma unroll
  for (int nt = 0; nt < 4; ++nt)
#pragma unroll
    for (int j = 0; j < 8; ++j) acc[nt][j] = 0.0f;

  for (int kb2 = 0; kb2 < 512; kb2 += 64) {
    __syncthreads();
    // ---- stage h1 tile (64 x 64 bf16, row stride 512) ----
#if defined(__gfx1250__) && __has_builtin(__builtin_amdgcn_tensor_load_to_lds)
    if (wid == 0) {
      const unsigned lds_off = (unsigned)(size_t)(void*)hs;  // generic addr low 32b = LDS offset
      const unsigned long long ga =
          (unsigned long long)(const void*)(h1 + (size_t)e0 * 512 + kb2);
      const unsigned rows = (unsigned)(EFF - e0);   // OOB rows read as zero
      u32x4 g0;
      g0[0] = 1u;                                    // count=1, user descriptor
      g0[1] = lds_off;                               // lds_addr
      g0[2] = (unsigned)(ga & 0xffffffffu);          // global_addr[31:0]
      g0[3] = (unsigned)((ga >> 32) & 0x1ffffffu) | (2u << 30);  // global_addr[56:32] | type=2
      i32x8 g1;
      g1[0] = (int)(1u << 16);                       // data_size = 2B
      g1[1] = (int)(512u << 16);                     // tensor_dim0[15:0] = 512
      g1[2] = (int)((rows & 0xffffu) << 16);         // tensor_dim0[31:16]=0 | tensor_dim1[15:0]
      g1[3] = (int)((rows >> 16) | (64u << 16));     // tensor_dim1[31:16] | tile_dim0 = 64
      g1[4] = (int)(64u);                            // tile_dim1 = 64, tile_dim2 = 0
      g1[5] = (int)(512u);                           // tensor_dim0_stride = 512
      g1[6] = 0;
      g1[7] = 0;
      i32x4 gz4;
      gz4[0] = 0; gz4[1] = 0; gz4[2] = 0; gz4[3] = 0;
      i32x8 gz8;
#pragma unroll
      for (int z = 0; z < 8; ++z) gz8[z] = 0;
      __builtin_amdgcn_tensor_load_to_lds(g0, g1, gz4, gz4, gz8, 0);
      __builtin_amdgcn_s_wait_tensorcnt(0);
    }
#else
#pragma unroll
    for (int t = 0; t < 4; ++t) {
      int idx = t * 128 + tid;
      int row = idx >> 3, c = idx & 7;
      int gge = e0 + row;
      uint4 val = make_uint4(0u, 0u, 0u, 0u);
      if (gge < EFF) val = ((const uint4*)(h1 + (size_t)gge * 512 + kb2))[c];
      ((uint4*)hs)[idx] = val;
    }
#endif
    for (int kk = 0; kk < 64; ++kk) {
      const int k = kb2 + kk;
      __syncthreads();
#pragma unroll
      for (int t = 0; t < 4; ++t) {                 // stage Wk2^T slice for this k (uint4 copies)
        int idx = t * 128 + tid;
        ((uint4*)bs)[idx] = ((const uint4*)(Wk2T + (size_t)k * 4096))[idx];
      }
      __builtin_prefetch(Wk2T + (size_t)(k + 1) * 4096, 0, 1);
      __syncthreads();
      const float hval = (float)hs[(wid * 16 + r15) * 64 + kk];
#pragma unroll
      for (int dh = 0; dh < 2; ++dh) {
        v16bf uf;                                    // B operand: u[d][e] = h[e,k]*x[e,d]
#pragma unroll
        for (int i = 0; i < 16; ++i) {
          int c = dh * 2 + (i >> 3);
          uf[i] = (bf16_t)(hval * xa[c * 8 + (i & 7)]);
        }
#pragma unroll
        for (int ft = 0; ft < 4; ++ft) {
          const v16bf af = frag_ld(&bs[(ft * 16 + r15) * 64 + dh * 32 + hh * 8]);
          acc[ft] = __builtin_amdgcn_wmma_f32_16x16x32_bf16(false, af, false, uf, (short)0,
                                                            acc[ft], false, false);
        }
      }
    }
  }

  // bias row (k = 512 holds bk2 transposed): h == 1
  __syncthreads();
#pragma unroll
  for (int t = 0; t < 4; ++t) {
    int idx = t * 128 + tid;
    ((uint4*)bs)[idx] = ((const uint4*)(Wk2T + (size_t)512 * 4096))[idx];
  }
  __syncthreads();
#pragma unroll
  for (int dh = 0; dh < 2; ++dh) {
    v16bf uf;
#pragma unroll
    for (int i = 0; i < 16; ++i) {
      int c = dh * 2 + (i >> 3);
      uf[i] = (bf16_t)xa[c * 8 + (i & 7)];
    }
#pragma unroll
    for (int ft = 0; ft < 4; ++ft) {
      const v16bf af = frag_ld(&bs[(ft * 16 + r15) * 64 + dh * 32 + hh * 8]);
      acc[ft] = __builtin_amdgcn_wmma_f32_16x16x32_bf16(false, af, false, uf, (short)0,
                                                        acc[ft], false, false);
    }
  }

  // scatter: lane owns one edge column
  if (didx >= 0) {
#pragma unroll
    for (int ft = 0; ft < 4; ++ft)
#pragma unroll
      for (int j = 0; j < 8; ++j)
        atomicAdd(&aggr[(size_t)didx * 64 + ft * 16 + j + hh * 8], acc[ft][j]);
  }
}

// ---------------- attention edge kernels ----------------
__global__ void k_score(const float* __restrict__ qg, const float* __restrict__ kg,
                        const float* __restrict__ ea_mf, const float* __restrict__ We,
                        const int* __restrict__ ei_mf, float* __restrict__ score,
                        unsigned* __restrict__ mkey) {
  const int lane = threadIdx.x & 31, wid = threadIdx.x >> 5;
  const int e = blockIdx.x * 8 + wid;
  if (e >= EMF) return;
  const int sm = ei_mf[e];
  const int df = ei_mf[EMF + e];
  float ea[6];
#pragma unroll
  for (int j = 0; j < 6; ++j) ea[j] = ea_mf[(size_t)e * 6 + j];
  float s = 0.0f;
  for (int c = lane; c < 64; c += 32) {
    float ke = kg[(size_t)sm * 64 + c];
#pragma unroll
    for (int j = 0; j < 6; ++j) ke += ea[j] * We[j * 64 + c];
    s += qg[(size_t)df * 64 + c] * ke;
  }
#pragma unroll
  for (int off = 16; off > 0; off >>= 1) s += __shfl_xor(s, off, 32);
  s *= 0.125f;
  if (lane == 0) {
    score[e] = s;
    unsigned b = __float_as_uint(s);
    unsigned key = (b & 0x80000000u) ? ~b : (b | 0x80000000u);
    atomicMax(&mkey[df], key);
  }
}

__global__ void k_expz(const int* __restrict__ ei_mf, const unsigned* __restrict__ mkey,
                       float* __restrict__ score, float* __restrict__ z) {
  int e = blockIdx.x * 256 + threadIdx.x;
  if (e >= EMF) return;
  const int df = ei_mf[EMF + e];
  unsigned key = mkey[df];
  unsigned b = (key & 0x80000000u) ? (key & 0x7fffffffu) : ~key;
  float m = __uint_as_float(b);
  float a = __expf(score[e] - m);
  score[e] = a;
  atomicAdd(&z[df], a);
}

__global__ void k_scatter_v(const int* __restrict__ ei_mf, const float* __restrict__ a,
                            const float* __restrict__ z, const float* __restrict__ vg,
                            float* __restrict__ xcr) {
  int idx = blockIdx.x * 256 + threadIdx.x;
  int e = idx >> 6, c = idx & 63;
  if (e >= EMF) return;
  const int sm = ei_mf[e];
  const int df = ei_mf[EMF + e];
  float alpha = a[e] / (z[df] + 1e-9f);
  atomicAdd(&xcr[(size_t)df * 64 + c], alpha * vg[(size_t)sm * 64 + c]);
}

// ---------------- final combine ----------------
__global__ void k_final(const float* __restrict__ rb, const float* __restrict__ broot,
                        const float* __restrict__ aggr, const float* __restrict__ xc2,
                        const float* __restrict__ bo, const float* __restrict__ ss,
                        float* __restrict__ out) {
  int i = blockIdx.x * 256 + threadIdx.x;
  if (i >= NF * 64) return;
  int d = i & 63;
  float t = rb[i] + broot[d] + aggr[i] + xc2[i] + bo[d];
  float g = fmaxf(t, 0.0f);
  float y = g * (1.0f + ss[d]) + ss[64 + d];
  out[i] = y / (1.0f + __expf(-y));
}

// ---------------- host orchestration ----------------
extern "C" void kernel_launch(void* const* d_in, const int* in_sizes, int n_in,
                              void* d_out, int out_size, void* d_ws, size_t ws_size,
                              hipStream_t stream) {
  const float* x_flow = (const float*)d_in[0];
  const float* x_memb = (const float*)d_in[1];
  const float* ea_ff  = (const float*)d_in[2];
  const float* ea_mf  = (const float*)d_in[3];
  const float* tau    = (const float*)d_in[4];
  const int*   ei_ff  = (const int*)d_in[5];
  const int*   ei_mf  = (const int*)d_in[6];
  const float* Wk0    = (const float*)d_in[7];
  const float* bk0    = (const float*)d_in[8];
  const float* Wk1    = (const float*)d_in[9];
  const float* bk1    = (const float*)d_in[10];
  const float* Wk2    = (const float*)d_in[11];
  const float* bk2    = (const float*)d_in[12];
  const float* Wroot  = (const float*)d_in[13];
  const float* broot  = (const float*)d_in[14];
  const float* Wq     = (const float*)d_in[15];
  const float* Wkk    = (const float*)d_in[16];
  const float* Wv     = (const float*)d_in[17];
  const float* We     = (const float*)d_in[18];
  const float* Wo     = (const float*)d_in[19];
  const float* bo     = (const float*)d_in[20];
  const float* Wt1    = (const float*)d_in[21];
  const float* bt1    = (const float*)d_in[22];
  const float* Wt2    = (const float*)d_in[23];
  const float* bt2    = (const float*)d_in[24];
  float* out = (float*)d_out;

  char* w = (char*)d_ws;
  size_t off = 0;
  auto alloc = [&](size_t bytes) -> void* {
    void* p = w + off;
    off = (off + bytes + 255) & ~(size_t)255;
    return p;
  };
  bf16_t* Wk1T   = (bf16_t*)alloc((size_t)512 * 512 * 2);
  bf16_t* Wk2T   = (bf16_t*)alloc((size_t)513 * 4096 * 2);   // row 512 = bk2 (transposed)
  bf16_t* WqT    = (bf16_t*)alloc(4096 * 2);
  bf16_t* WkkT   = (bf16_t*)alloc(4096 * 2);
  bf16_t* WvT    = (bf16_t*)alloc(4096 * 2);
  bf16_t* WrootT = (bf16_t*)alloc(4096 * 2);
  bf16_t* WoT    = (bf16_t*)alloc(4096 * 2);
  float*  ssb    = (float*)alloc(128 * 4);
  float*  qb     = (float*)alloc((size_t)NF * 64 * 4);
  float*  kbuf   = (float*)alloc((size_t)NM * 64 * 4);
  float*  vbuf   = (float*)alloc((size_t)NM * 64 * 4);
  float*  rbuf   = (float*)alloc((size_t)NF * 64 * 4);
  float*  scoreb = (float*)alloc((size_t)EMF * 4);
  unsigned* mkeyb = (unsigned*)alloc((size_t)NF * 4);
  float*  zbuf   = (float*)alloc((size_t)NF * 4);
  float*  xcrb   = (float*)alloc((size_t)NF * 64 * 4);
  float*  xc2b   = (float*)alloc((size_t)NF * 64 * 4);
  float*  aggrb  = (float*)alloc((size_t)NF * 64 * 4);
  bf16_t* h1b    = (bf16_t*)alloc((size_t)EFF * 512 * 2);
  (void)ws_size; (void)n_in; (void)in_sizes; (void)out_size;

  auto zero = [&](float* p, int n) {
    k_zero_f32<<<(n + 255) / 256, 256, 0, stream>>>(p, n);
  };

  // 1) weight conversions (+ transposes so all WMMA fragments are K-contiguous)
  k_cvt_t512<<<(512 * 512 + 255) / 256, 256, 0, stream>>>(Wk1, Wk1T);
  k_cvt_tk2<<<(512 * 4096 + 255) / 256, 256, 0, stream>>>(Wk2, Wk2T);
  k_cvt_tb2<<<16, 256, 0, stream>>>(bk2, Wk2T);
  k_cvt_t64<<<16, 256, 0, stream>>>(Wq, WqT);
  k_cvt_t64<<<16, 256, 0, stream>>>(Wkk, WkkT);
  k_cvt_t64<<<16, 256, 0, stream>>>(Wv, WvT);
  k_cvt_t64<<<16, 256, 0, stream>>>(Wroot, WrootT);
  k_cvt_t64<<<16, 256, 0, stream>>>(Wo, WoT);

  // 2) FiLM params
  k_temb<<<1, 128, 0, stream>>>(tau, Wt1, bt1, Wt2, bt2, ssb);

  // 3) zero accumulators
  zero((float*)mkeyb, NF);
  zero(zbuf, NF);
  zero(xcrb, NF * 64);
  zero(aggrb, NF * 64);

  // 4) 64x64 projections (WMMA)
  k_gemm64_wmma<<<(NF + 127) / 128, 256, 0, stream>>>(x_flow, WqT, qb, NF);
  k_gemm64_wmma<<<(NM + 127) / 128, 256, 0, stream>>>(x_memb, WkkT, kbuf, NM);
  k_gemm64_wmma<<<(NM + 127) / 128, 256, 0, stream>>>(x_memb, WvT, vbuf, NM);
  k_gemm64_wmma<<<(NF + 127) / 128, 256, 0, stream>>>(x_flow, WrootT, rbuf, NF);

  // 5) cross attention (segment softmax via atomics)
  k_score<<<(EMF + 7) / 8, 256, 0, stream>>>(qb, kbuf, ea_mf, We, ei_mf, scoreb, mkeyb);
  k_expz<<<(EMF + 255) / 256, 256, 0, stream>>>(ei_mf, mkeyb, scoreb, zbuf);
  k_scatter_v<<<(EMF * 64 + 255) / 256, 256, 0, stream>>>(ei_mf, scoreb, zbuf, vbuf, xcrb);
  k_gemm64_wmma<<<(NF + 127) / 128, 256, 0, stream>>>(xcrb, WoT, xc2b, NF);

  // 6) fused edge MLP
  k_h1_wmma<<<dim3((EFF + 127) / 128, 8), 256, 0, stream>>>(ea_ff, Wk0, bk0, Wk1T, bk1, h1b);

  // 7) edge-conditioned message GEMM + scatter (the 419 GFLOP step)
  k_msg_wmma<<<(EFF + 63) / 64, 128, 0, stream>>>(x_flow, ei_ff, Wk2T, h1b, aggrb);

  // 8) final combine
  k_final<<<(NF * 64 + 255) / 256, 256, 0, stream>>>(rbuf, broot, aggrb, xc2b, bo, ssb, out);
}
